// SpGraphAttentionLayer_58935541235965
// MI455X (gfx1250) — matrix operand order repaired and checked by reading, not compiled, not run on hardware
//
#include <hip/hip_runtime.h>
#include <hip/hip_bf16.h>

#define M_NODES 50000
#define K_F     256
#define N_F     128
#define N_EDGES 800000
#define N_RELS  50

typedef __attribute__((ext_vector_type(16))) _Float16 v16h;
typedef __attribute__((ext_vector_type(8)))  _Float16 v8h;
typedef __attribute__((ext_vector_type(8)))  float    v8f;

#define LDS_PITCH 264   // 256 + 8 halfs: 528B row pitch -> conflict-free column reads

// ---------------------------------------------------------------------------
// h = input @ W  via v_wmma_f32_16x16x32_f16.
// Block = 256 threads = 8 waves. blockIdx.x = M-tile (50000/16 = 3125 exact);
// wave id = N-tile (128/16 = 8 exact). The 16x256 A tile is staged once in
// LDS as f16 (co-op load by all 8 waves), then each wave runs 8 fully
// unrolled WMMA steps reading A fragments from LDS (ds_load_b128) and its
// private 16-column W slice from global (L2-resident, read exactly once
// per block).
// ---------------------------------------------------------------------------
__global__ __launch_bounds__(256) void gemm_h_wmma(
    const float* __restrict__ A,   // [M,K] row-major (input)
    const float* __restrict__ W,   // [K,N] row-major
    float* __restrict__ H)         // [M,N]
{
    __shared__ __align__(16) _Float16 As[16 * LDS_PITCH];

    const int m0 = blockIdx.x * 16;

    // Cooperative A-tile stage: 256 threads x 16 floats = 16x256 tile.
    {
        const int row = threadIdx.x >> 4;         // 0..15
        const int c0  = (threadIdx.x & 15) * 16;  // 0,16,...,240
        const float* srcp = A + (size_t)(m0 + row) * K_F + c0;
        _Float16*    dstp = As + row * LDS_PITCH + c0;
        #pragma unroll
        for (int j = 0; j < 16; ++j) dstp[j] = (_Float16)srcp[j];
    }
    __syncthreads();

    const int lane = threadIdx.x & 31;
    const int wave = threadIdx.x >> 5;      // 0..7  -> N tile
    const int n0   = wave * 16;
    const int g    = lane >> 4;             // lane group (0: lanes 0-15, 1: 16-31)
    const int l    = lane & 15;

    v8f acc = {};
    const _Float16* arow = As + l * LDS_PITCH;

    #pragma unroll
    for (int k0 = 0; k0 < K_F; k0 += 32) {
        // A 16x32 f16 fragment: group 0 holds K {0..7,16..23}, group 1 {8..15,24..31}
        const int ka = k0 + g * 8;
        v8h alo = *(const v8h*)(arow + ka);        // ds_load_b128
        v8h ahi = *(const v8h*)(arow + ka + 16);   // ds_load_b128
        v16h a  = __builtin_shufflevector(alo, ahi,
                    0, 1, 2, 3, 4, 5, 6, 7, 8, 9, 10, 11, 12, 13, 14, 15);

        // B 32x16 f16 fragment: lane column n0+l; group 0 rows K {0..7,16..23}, group 1 {8..15,24..31}
        v16h b;
        const float* bcol = W + (size_t)(k0 + g * 8) * N_F + n0 + l;
        #pragma unroll
        for (int j = 0; j < 16; ++j) b[j] = (_Float16)bcol[(size_t)(j + (j & 8)) * N_F];

        acc = __builtin_amdgcn_wmma_f32_16x16x32_f16(
            false, a, false, b, (short)0, acc, false, false);
    }

    // C/D layout: VGPR i holds row M = i + 8*g, column N = l
    float* out = H + (size_t)m0 * N_F + n0 + l;
    #pragma unroll
    for (int i = 0; i < 8; ++i)
        out[(size_t)(i + g * 8) * N_F] = acc[i];
}

// ---------------------------------------------------------------------------
// ra2[r] = rel_emb[r] . a2   (50 x 128 dots) — one wave per relation
// ---------------------------------------------------------------------------
__global__ void rel_scores(const float* __restrict__ rel_emb,
                           const float* __restrict__ a,   // [2*N_F], a2 = a+N_F
                           float* __restrict__ ra2)
{
    const int r    = blockIdx.x;
    const int lane = threadIdx.x & 31;
    const float* rr = rel_emb + (size_t)r * N_F;
    float d = 0.f;
    #pragma unroll
    for (int j = 0; j < 4; ++j) {
        const int f = lane * 4 + j;
        d += rr[f] * a[N_F + f];
    }
    #pragma unroll
    for (int m = 16; m >= 1; m >>= 1) d += __shfl_xor(d, m, 32);
    if (lane == 0) ra2[r] = d;
}

// ---------------------------------------------------------------------------
// Per node: u = h.a1, v = h.a2, e_eye = exp(-leaky(u+v));
// initialize rowsum = e_eye and h' (= d_out) = e_eye * h  (self-loop term).
// One wave per node, 4 features per lane. Re-initializes all accumulators
// each call -> deterministic under graph replay.
// ---------------------------------------------------------------------------
__global__ __launch_bounds__(256) void node_init(
    const float* __restrict__ H,
    const float* __restrict__ a,        // [2*N_F]
    float* __restrict__ u, float* __restrict__ v,
    float* __restrict__ rowsum,
    float* __restrict__ hprime)         // == d_out
{
    const int node = blockIdx.x * 8 + (threadIdx.x >> 5);
    if (node >= M_NODES) return;
    const int lane = threadIdx.x & 31;
    const float* hr = H + (size_t)node * N_F;

    float hv[4], du = 0.f, dv = 0.f;
    #pragma unroll
    for (int j = 0; j < 4; ++j) {
        const int f = lane * 4 + j;
        hv[j] = hr[f];
        du += hv[j] * a[f];
        dv += hv[j] * a[N_F + f];
    }
    #pragma unroll
    for (int m = 16; m >= 1; m >>= 1) {
        du += __shfl_xor(du, m, 32);
        dv += __shfl_xor(dv, m, 32);
    }
    const float s  = du + dv;
    const float ls = (s >= 0.f) ? s : 0.2f * s;
    const float ee = __expf(-ls);

    if (lane == 0) { u[node] = du; v[node] = dv; rowsum[node] = ee; }
    float* pr = hprime + (size_t)node * N_F + lane * 4;
    #pragma unroll
    for (int j = 0; j < 4; ++j) pr[j] = ee * hv[j];
}

// ---------------------------------------------------------------------------
// Edge scatter: s_fwd = u[src]+v[dst]-ra2[rel], s_bwd = u[dst]+v[src]+ra2[rel].
// e = exp(-leaky(s)). Atomically accumulate rowsums and e * h[other] rows.
// One wave per edge; h/h' both fit in the 192MB L2 so the f32 hardware
// atomics (global_atomic_add_f32) resolve on-chip.
// ---------------------------------------------------------------------------
__global__ __launch_bounds__(256) void edge_scatter(
    const float* __restrict__ H,
    const int* __restrict__ src, const int* __restrict__ dst,
    const int* __restrict__ rel,
    const float* __restrict__ u, const float* __restrict__ v,
    const float* __restrict__ ra2,
    float* __restrict__ rowsum, float* __restrict__ hprime)
{
    const int e = blockIdx.x * 8 + (threadIdx.x >> 5);
    if (e >= N_EDGES) return;
    const int lane = threadIdx.x & 31;

    const int s = src[e];
    const int d = dst[e];
    const int r = rel[e];

    const float rd = ra2[r];
    const float sf = u[s] + v[d] - rd;
    const float sb = u[d] + v[s] + rd;
    const float lf = (sf >= 0.f) ? sf : 0.2f * sf;
    const float lb = (sb >= 0.f) ? sb : 0.2f * sb;
    const float ef = __expf(-lf);
    const float eb = __expf(-lb);

    if (lane == 0) {
        unsafeAtomicAdd(&rowsum[s], ef);
        unsafeAtomicAdd(&rowsum[d], eb);
    }

    const float* hs = H + (size_t)s * N_F;
    const float* hd = H + (size_t)d * N_F;
    float* ps = hprime + (size_t)s * N_F;
    float* pd = hprime + (size_t)d * N_F;
    #pragma unroll
    for (int j = 0; j < 4; ++j) {
        const int f = lane * 4 + j;
        unsafeAtomicAdd(&ps[f], ef * hd[f]);
        unsafeAtomicAdd(&pd[f], eb * hs[f]);
    }
}

// ---------------------------------------------------------------------------
// Finalize in place: out = elu(h' / rowsum)
// ---------------------------------------------------------------------------
__global__ void finalize(float* __restrict__ out,
                         const float* __restrict__ rowsum)
{
    const size_t i = (size_t)blockIdx.x * blockDim.x + threadIdx.x;
    if (i >= (size_t)M_NODES * N_F) return;
    const int node = (int)(i >> 7);          // N_F == 128
    const float x = out[i] / rowsum[node];
    out[i] = (x > 0.f) ? x : (__expf(x) - 1.f);
}

extern "C" void kernel_launch(void* const* d_in, const int* in_sizes, int n_in,
                              void* d_out, int out_size, void* d_ws, size_t ws_size,
                              hipStream_t stream)
{
    (void)in_sizes; (void)n_in; (void)out_size; (void)ws_size;

    const float* input   = (const float*)d_in[0];   // [50000,256]
    const float* rel_emb = (const float*)d_in[1];   // [50,128]
    const float* W       = (const float*)d_in[2];   // [256,128]
    const float* a       = (const float*)d_in[3];   // [1,256]
    const int*   eidx    = (const int*)d_in[4];     // [2,800000]
    const int*   erel    = (const int*)d_in[5];     // [800000]

    const int* src = eidx;
    const int* dst = eidx + N_EDGES;

    // Workspace layout (floats): h | u | v | rowsum | ra2
    float* h      = (float*)d_ws;
    float* u      = h + (size_t)M_NODES * N_F;
    float* v      = u + M_NODES;
    float* rowsum = v + M_NODES;
    float* ra2    = rowsum + M_NODES;

    float* out = (float*)d_out;   // doubles as h' accumulator

    // 1) h = input @ W  (WMMA, LDS-staged A tile)
    gemm_h_wmma<<<M_NODES / 16, 256, 0, stream>>>(input, W, h);

    // 2) relation score precompute
    rel_scores<<<N_RELS, 32, 0, stream>>>(rel_emb, a, ra2);

    // 3) per-node scores + self-loop init of accumulators
    node_init<<<(M_NODES + 7) / 8, 256, 0, stream>>>(h, a, u, v, rowsum, out);

    // 4) edge scatter (dominant phase; L2-resident atomics)
    edge_scatter<<<(N_EDGES + 7) / 8, 256, 0, stream>>>(h, src, dst, erel,
                                                        u, v, ra2, rowsum, out);

    // 5) normalize + ELU in place
    const size_t total = (size_t)M_NODES * N_F;
    finalize<<<(unsigned)((total + 255) / 256), 256, 0, stream>>>(out, rowsum);
}
